// FrequencyLocalMHSA_7619271983795
// MI455X (gfx1250) — matrix operand order repaired and checked by reading, not compiled, not run on hardware
//
#include <hip/hip_runtime.h>
#include <hip/hip_bf16.h>

typedef __attribute__((ext_vector_type(16))) _Float16 v16h;
typedef __attribute__((ext_vector_type(8)))  _Float16 v8h;
typedef __attribute__((ext_vector_type(8)))  float    v8f;
typedef __attribute__((ext_vector_type(4)))  float    v4f;

#define NEG_BIG (-3.0e38f)

__device__ __forceinline__ v16h cat16(v8h lo, v8h hi) {
  return __builtin_shufflevector(lo, hi, 0,1,2,3,4,5,6,7,8,9,10,11,12,13,14,15);
}
__device__ __forceinline__ v8f wmma_f16(v16h a, v16h b, v8f c) {
  // D = A(16x32 f16) * B(32x16 f16) + C(16x16 f32)
  return __builtin_amdgcn_wmma_f32_16x16x32_f16(false, a, false, b, (short)0, c, false, false);
}

// CDNA5 async copy global->LDS (GVS mode: mem = saddr64 + vaddr32 + ioffset),
// 16B per lane, tracked by ASYNCcnt. Compiler cannot see these, so every
// consumer barrier is preceded by an explicit s_wait_asynccnt 0.
__device__ __forceinline__ void async_ld_b128(void* lds, unsigned long long gbase,
                                              unsigned goff) {
  unsigned laddr = (unsigned)(unsigned long long)lds;  // low 32 bits = LDS offset
  asm volatile("global_load_async_to_lds_b128 %0, %1, %2"
               :: "v"(laddr), "v"(goff), "s"(gbase) : "memory");
}
__device__ __forceinline__ void wait_async0() {
  asm volatile("s_wait_asynccnt 0x0" ::: "memory");
}

// ---------------------------------------------------------------------------
// Weight conversion: w_in [768,256] f32 -> f16 (already B^T layout: row n,
// contiguous k), w_out [256,256] f32 -> f16.
// ---------------------------------------------------------------------------
__global__ __launch_bounds__(256) void cvt_weights_kernel(
    const float* __restrict__ w_in, const float* __restrict__ w_out,
    _Float16* __restrict__ wh) {
  int idx = blockIdx.x * 256 + threadIdx.x;   // grid covers 262144
  if (idx < 196608) wh[idx] = (_Float16)w_in[idx];
  else              wh[idx] = (_Float16)w_out[idx - 196608];
}

// ---------------------------------------------------------------------------
// Tiled GEMM + bias: out[m][n] = sum_k A[m][k] * Bt[n][k] + bias[n]
// M=65536, K=256. Block tile 128(M) x 128(N), 256 threads = 8 waves,
// each wave owns 16 M-rows x 128 N-cols (8 WMMA accumulators = 64 VGPRs).
// Double-buffered LDS; f16 operands staged with async-to-LDS.
// ---------------------------------------------------------------------------
template<typename AT, bool OUT_HALF>
__global__ __launch_bounds__(256) void gemm_bias_kernel(
    const AT* __restrict__ A, const _Float16* __restrict__ Bt,
    const float* __restrict__ bias, void* __restrict__ outp, int Nstride)
{
  __shared__ _Float16 As[2][128][40];   // 32 K + 8 pad (keeps 16B alignment)
  __shared__ _Float16 Bs[2][128][40];

  const int tid  = threadIdx.x;
  const int lane = tid & 31;
  const int wid  = tid >> 5;
  const int hi   = (lane >> 4) & 1;
  const int l15  = lane & 15;
  const int m0   = blockIdx.x * 128;
  const int n0   = blockIdx.y * 128;

  const int r    = tid >> 1;          // staging row (0..127)
  const int cseg = (tid & 1) * 16;    // staging K-half

  const unsigned long long bbase = (unsigned long long)Bt;
  const unsigned long long abase = (unsigned long long)A;

  auto stage = [&](int buf, int k0) {
    // B^T tile 128x32 f16: two async b128 per thread
    {
      const unsigned goff = (unsigned)(((n0 + r) * 256 + k0 + cseg) * 2);
      async_ld_b128(&Bs[buf][r][cseg],     bbase, goff);
      async_ld_b128(&Bs[buf][r][cseg + 8], bbase, goff + 16u);
    }
    if constexpr (sizeof(AT) == 4) {
      // A tile 128x32 f32 -> f16 (manual: async cannot convert)
      const float* src = (const float*)A + (size_t)(m0 + r) * 256 + k0 + cseg;
      #pragma unroll
      for (int u = 0; u < 16; u += 4) {
        v4f f = *(const v4f*)(src + u);
        As[buf][r][cseg + u + 0] = (_Float16)f.x;
        As[buf][r][cseg + u + 1] = (_Float16)f.y;
        As[buf][r][cseg + u + 2] = (_Float16)f.z;
        As[buf][r][cseg + u + 3] = (_Float16)f.w;
      }
    } else {
      const unsigned goff = (unsigned)(((m0 + r) * 256 + k0 + cseg) * 2);
      async_ld_b128(&As[buf][r][cseg],     abase, goff);
      async_ld_b128(&As[buf][r][cseg + 8], abase, goff + 16u);
    }
  };

  v8f acc[8] = {};

  stage(0, 0);
  wait_async0();
  __syncthreads();

  for (int ki = 0; ki < 8; ki++) {
    const int buf = ki & 1;
    if (ki < 7) stage(buf ^ 1, (ki + 1) * 32);   // prefetch next tile (async)

    // A fragment: lane l15 = row, VGPRs 0-3: K=(hi?8:0)..+7, 4-7: K=16+(hi?8:0)..+7
    const int mA = wid * 16 + l15;
    v16h a = cat16(*(const v8h*)&As[buf][mA][hi * 8],
                   *(const v8h*)&As[buf][mA][16 + hi * 8]);
    #pragma unroll
    for (int nt = 0; nt < 8; nt++) {
      // B fragment: lane l15 = col, VGPRs 0-3: K=(hi?8:0)..+7, 4-7: K=16+(hi?8:0)..+7
      const int nB = nt * 16 + l15;
      v16h b = cat16(*(const v8h*)&Bs[buf][nB][hi * 8],
                     *(const v8h*)&Bs[buf][nB][16 + hi * 8]);
      acc[nt] = wmma_f16(a, b, acc[nt]);
    }
    wait_async0();     // next buffer's async copies complete
    __syncthreads();   // all waves done reading current buffer
  }

  // Epilogue: C/D layout lane l15 = col n, VGPR v = row (v + 8*hi)
  #pragma unroll
  for (int nt = 0; nt < 8; nt++) {
    const int   ng = n0 + nt * 16 + l15;
    const float bv = bias[ng];
    #pragma unroll
    for (int v = 0; v < 8; v++) {
      const int   mg  = m0 + wid * 16 + v + hi * 8;
      const float val = acc[nt][v] + bv;
      if constexpr (OUT_HALF)
        ((_Float16*)outp)[(size_t)mg * Nstride + ng] = (_Float16)val;
      else
        ((float*)outp)[(size_t)mg * Nstride + ng] = val;
    }
  }
}

// ---------------------------------------------------------------------------
// Banded attention, one workgroup per (bt, h). 256 threads = 8 waves,
// each wave processes two 16-row blocks. Scores/out both via WMMA.
// qkv layout: [65536][768] f16, q cols [0,256), k [256,512), v [512,768).
// ---------------------------------------------------------------------------
__global__ __launch_bounds__(256) void attn_kernel(
    const _Float16* __restrict__ qkv, _Float16* __restrict__ aout,
    const int* __restrict__ band_p)
{
  __shared__ _Float16 Ks[256][72];      // k rows (f, d)      36.9 KB
  __shared__ _Float16 Vt[64][320];      // v^T: (d, 16+j)     41.0 KB
  __shared__ _Float16 Pn[8][16][72];    // per-wave attn f16  18.4 KB

  const int tid  = threadIdx.x;
  const int lane = tid & 31;
  const int wid  = tid >> 5;
  const int hi   = (lane >> 4) & 1;
  const int l15  = lane & 15;
  const int bt   = blockIdx.x;
  const int h    = blockIdx.y;
  const int band = *band_p;

  // Kick off async copy of all K rows into LDS (8x b128 per thread; row = tid)
  {
    const unsigned long long kbase =
        (unsigned long long)(qkv + (size_t)bt * 256 * 768 + 256 + h * 64);
    const unsigned goff = (unsigned)(tid * 768 * 2);
    char* ldsrow = (char*)&Ks[tid][0];
    #pragma unroll
    for (int u = 0; u < 128; u += 16)
      async_ld_b128(ldsrow + u, kbase, goff + (unsigned)u);
  }

  // Zero Vt (guard columns must be 0 so masked attn never meets NaN garbage)
  {
    _Float16* p = &Vt[0][0] + tid * 80;   // 64*320/256 = 80 halves per thread
    v8h z = {};
    #pragma unroll
    for (int u = 0; u < 80; u += 8) *(v8h*)(p + u) = z;
  }
  __syncthreads();

  // Stage V transposed (with 16-col left guard)
  {
    const int f = tid;
    const _Float16* vrow = qkv + (size_t)(bt * 256 + f) * 768 + 512 + h * 64;
    #pragma unroll
    for (int u = 0; u < 64; u += 8) {
      v8h vv = *(const v8h*)(vrow + u);
      #pragma unroll
      for (int e = 0; e < 8; e++) Vt[u + e][16 + f] = vv[e];
    }
  }
  wait_async0();     // K rows landed in LDS
  __syncthreads();

  for (int rb = 0; rb < 2; rb++) {
    const int i0 = (wid + rb * 8) * 16;

    // Q fragments straight from global (row = i0 + l15, 64 d-values)
    const _Float16* qrow = qkv + (size_t)(bt * 256 + i0 + l15) * 768 + h * 64;
    v16h qa0 = cat16(*(const v8h*)(qrow + hi * 8),      *(const v8h*)(qrow + 16 + hi * 8));
    v16h qa1 = cat16(*(const v8h*)(qrow + 32 + hi * 8), *(const v8h*)(qrow + 48 + hi * 8));

    // Scores: 3 column tiles j0 = i0-16, i0, i0+16
    float p[3][8];
    float rmax[8];
    #pragma unroll
    for (int v = 0; v < 8; v++) rmax[v] = NEG_BIG;

    #pragma unroll
    for (int jt = 0; jt < 3; jt++) {
      const int j0 = i0 + (jt - 1) * 16;
      if (j0 < 0 || j0 >= 256) {
        #pragma unroll
        for (int v = 0; v < 8; v++) p[jt][v] = NEG_BIG;
        continue;
      }
      const int nK = j0 + l15;
      v16h b0 = cat16(*(const v8h*)&Ks[nK][hi * 8],      *(const v8h*)&Ks[nK][16 + hi * 8]);
      v16h b1 = cat16(*(const v8h*)&Ks[nK][32 + hi * 8], *(const v8h*)&Ks[nK][48 + hi * 8]);
      v8f s = {};
      s = wmma_f16(qa0, b0, s);
      s = wmma_f16(qa1, b1, s);
      const int j = j0 + l15;
      #pragma unroll
      for (int v = 0; v < 8; v++) {
        const int i = i0 + v + hi * 8;
        int dd = i - j; if (dd < 0) dd = -dd;
        float sv = s[v] * 0.125f;           // 1/sqrt(64)
        p[jt][v] = (dd <= band) ? sv : NEG_BIG;
        rmax[v]  = fmaxf(rmax[v], p[jt][v]);
      }
    }

    // Softmax: rows live in 16-lane halves; xor masks 1..8 stay within a half
    #pragma unroll
    for (int v = 0; v < 8; v++) {
      float m = rmax[v];
      #pragma unroll
      for (int mask = 1; mask < 16; mask <<= 1)
        m = fmaxf(m, __shfl_xor(m, mask, 32));
      float sum = 0.f;
      #pragma unroll
      for (int jt = 0; jt < 3; jt++) {
        float e = __expf(p[jt][v] - m);
        p[jt][v] = e;
        sum += e;
      }
      #pragma unroll
      for (int mask = 1; mask < 16; mask <<= 1)
        sum += __shfl_xor(sum, mask, 32);
      const float inv = 1.0f / sum;
      #pragma unroll
      for (int jt = 0; jt < 3; jt++) p[jt][v] *= inv;
    }

    // Stage attn into per-wave LDS scratch (zero cols 0..63 first)
    {
      _Float16* pz = &Pn[wid][l15][hi * 32];
      v8h z = {};
      #pragma unroll
      for (int u = 0; u < 32; u += 8) *(v8h*)(pz + u) = z;
    }
    #pragma unroll
    for (int jt = 0; jt < 3; jt++)
      #pragma unroll
      for (int v = 0; v < 8; v++)
        Pn[wid][v + hi * 8][jt * 16 + l15] = (_Float16)p[jt][v];

    // out[i][d] = attn[i][k] * V[jb+k][d],  jb = i0-16, Vt col = 16+jb+k = i0+k
    v16h pa0 = cat16(*(const v8h*)&Pn[wid][l15][hi * 8],
                     *(const v8h*)&Pn[wid][l15][16 + hi * 8]);
    v16h pa1 = cat16(*(const v8h*)&Pn[wid][l15][32 + hi * 8],
                     *(const v8h*)&Pn[wid][l15][48 + hi * 8]);
    #pragma unroll
    for (int nt = 0; nt < 4; nt++) {
      const int n = nt * 16 + l15;  // d index
      v16h b0 = cat16(*(const v8h*)&Vt[n][i0 + hi * 8],
                      *(const v8h*)&Vt[n][i0 + 16 + hi * 8]);
      v16h b1 = cat16(*(const v8h*)&Vt[n][i0 + 32 + hi * 8],
                      *(const v8h*)&Vt[n][i0 + 48 + hi * 8]);
      v8f c = {};
      c = wmma_f16(pa0, b0, c);
      c = wmma_f16(pa1, b1, c);
      const int dcol = nt * 16 + l15;
      #pragma unroll
      for (int v = 0; v < 8; v++) {
        const int frow = i0 + v + hi * 8;
        aout[(size_t)(bt * 256 + frow) * 256 + h * 64 + dcol] = (_Float16)c[v];
      }
    }
  }
}

// ---------------------------------------------------------------------------
extern "C" void kernel_launch(void* const* d_in, const int* in_sizes, int n_in,
                              void* d_out, int out_size, void* d_ws, size_t ws_size,
                              hipStream_t stream) {
  const float* x     = (const float*)d_in[0];   // [65536][256]
  const float* w_in  = (const float*)d_in[1];   // [768][256]
  const float* b_in  = (const float*)d_in[2];   // [768]
  const float* w_out = (const float*)d_in[3];   // [256][256]
  const float* b_out = (const float*)d_in[4];   // [256]
  const int*   band  = (const int*)d_in[5];

  _Float16* wh      = (_Float16*)d_ws;
  _Float16* w_in_h  = wh;                              // 196608 halves
  _Float16* w_out_h = wh + 196608;                     //  65536 halves
  _Float16* qkv     = wh + 262144;                     // [65536][768]
  _Float16* aouth   = qkv + (size_t)65536 * 768;       // [65536][256]

  cvt_weights_kernel<<<1024, 256, 0, stream>>>(w_in, w_out, wh);
  gemm_bias_kernel<float, true><<<dim3(512, 6), 256, 0, stream>>>(
      x, w_in_h, b_in, (void*)qkv, 768);
  attn_kernel<<<dim3(256, 4), 256, 0, stream>>>(qkv, aouth, band);
  gemm_bias_kernel<_Float16, false><<<dim3(512, 2), 256, 0, stream>>>(
      aouth, w_out_h, b_out, d_out, 256);
}
